// GCN_58686433132688
// MI455X (gfx1250) — matrix operand order repaired and checked by reading, not compiled, not run on hardware
//
#include <hip/hip_runtime.h>
#include <math.h>

// ---------------------------------------------------------------------------
// 4-layer GCN for MI455X (gfx1250, wave32).
//   per layer:  h = cur @ W          (wave-per-16x16-tile, v_wmma_f32_16x16x32_f16)
//               out = b + h*dinv^2   (self-loop folded into init)
//               out[col] += h[row]*norm   (wave-per-edge, hw float atomics)
//               leaky_relu (layers 1..3)
// norm/dinv computed once (layer-invariant).
// ---------------------------------------------------------------------------

typedef __attribute__((ext_vector_type(16))) _Float16 v16h;
typedef __attribute__((ext_vector_type(8)))  float    v8f;

#define NEG_SLOPE 0.01f

// ---------------- gcn_norm precompute ----------------

__global__ __launch_bounds__(256) void k_deg_init(float* __restrict__ deg, int n) {
  int i = blockIdx.x * blockDim.x + threadIdx.x;
  if (i < n) deg[i] = 1.0f;  // self-loop weight
}

__global__ __launch_bounds__(256) void k_deg_acc(const int* __restrict__ cols,
                                                 const float* __restrict__ w,
                                                 float* __restrict__ deg, int e) {
  int i = blockIdx.x * blockDim.x + threadIdx.x;
  if (i < e) unsafeAtomicAdd(&deg[cols[i]], w[i]);
}

__global__ __launch_bounds__(256) void k_dinv(float* __restrict__ d, int n) {
  int i = blockIdx.x * blockDim.x + threadIdx.x;
  if (i < n) { float v = d[i]; d[i] = (v > 0.0f) ? rsqrtf(v) : 0.0f; }
}

__global__ __launch_bounds__(256) void k_norm(const int* __restrict__ rows,
                                              const int* __restrict__ cols,
                                              const float* __restrict__ w,
                                              const float* __restrict__ dinv,
                                              float* __restrict__ norm, int e) {
  int i = blockIdx.x * blockDim.x + threadIdx.x;
  if (i < e) norm[i] = dinv[rows[i]] * w[i] * dinv[cols[i]];
}

// ---------------- dense GEMM via WMMA ----------------
// H[N x COUT] = X[N x 128] * W[128 x COUT], fp32 in/out, f16 multiply, f32 acc.
// One wave computes one 16x16 tile of H. nrows must be a multiple of 16
// (N = 50000 = 3125*16).

template <int COUT>
__global__ __launch_bounds__(128) void k_gemm_wmma(const float* __restrict__ X,
                                                   const float* __restrict__ W,
                                                   float* __restrict__ H, int nrows) {
  const int lane = threadIdx.x & 31;
  const int wave = threadIdx.x >> 5;
  const int tiles_n = COUT / 16;
  const int total_tiles = (nrows / 16) * tiles_n;
  const int tile = blockIdx.x * 4 + wave;
  if (tile >= total_tiles) return;
  const int mt = tile / tiles_n;
  const int nt = tile % tiles_n;

  const int half = lane >> 4;   // 0: lanes 0-15, 1: lanes 16-31
  const int l16  = lane & 15;

  const float* xrow = X + (size_t)(mt * 16 + l16) * 128;       // A: row = l16
  const float* wcol = W + (size_t)(nt * 16 + l16);             // B: col = l16

  v8f c = {};
#pragma unroll
  for (int k0 = 0; k0 < 128; k0 += 32) {
    // A fragment (16x32 f16): elems 0..7 -> K = k0+half*8+{0..7},
    //                         elems 8..15 -> K = k0+16+half*8+{0..7}
    v16h a, b;
    const int ka = k0 + half * 8;
#pragma unroll
    for (int v = 0; v < 4; ++v) {
      a[2 * v]     = (_Float16)xrow[ka + 2 * v];
      a[2 * v + 1] = (_Float16)xrow[ka + 2 * v + 1];
      a[8 + 2 * v]     = (_Float16)xrow[ka + 16 + 2 * v];
      a[8 + 2 * v + 1] = (_Float16)xrow[ka + 16 + 2 * v + 1];
    }
    // B fragment (32x16 f16): same K order as A, col = l16
    const int kb = k0 + half * 8;
#pragma unroll
    for (int j = 0; j < 16; ++j) {
      b[j] = (_Float16)wcol[(size_t)(kb + j + (j >= 8 ? 8 : 0)) * COUT];
    }
    c = __builtin_amdgcn_wmma_f32_16x16x32_f16(false, a, false, b,
                                               (short)0, c, false, false);
  }
  // D layout: lane holds col = nt*16 + l16, rows = mt*16 + half*8 + v
  float* hdst = H + (size_t)(mt * 16 + half * 8) * COUT + nt * 16 + l16;
#pragma unroll
  for (int v = 0; v < 8; ++v) {
    hdst[(size_t)v * COUT] = c[v];
  }
}

// ---------------- out = bias + h * dinv^2 (self-loop) ----------------

template <int COUT>
__global__ __launch_bounds__(256) void k_init_out(const float* __restrict__ H,
                                                  const float* __restrict__ dinv,
                                                  const float* __restrict__ bias,
                                                  float* __restrict__ out, int n) {
  int idx = blockIdx.x * blockDim.x + threadIdx.x;
  if (idx >= n * COUT) return;
  const int node = idx / COUT;
  const int ch   = idx % COUT;
  const float di = dinv[node];
  out[idx] = bias[ch] + H[idx] * di * di;
}

// ---------------- edge scatter: out[col] += h[row] * norm ----------------
// One wave32 per edge; each lane owns COUT/32 contiguous channels.

template <int COUT>
__global__ __launch_bounds__(256) void k_scatter(const int* __restrict__ rows,
                                                 const int* __restrict__ cols,
                                                 const float* __restrict__ norm,
                                                 const float* __restrict__ H,
                                                 float* __restrict__ out, int e) {
  const int eidx = blockIdx.x * 8 + (threadIdx.x >> 5);
  if (eidx >= e) return;
  const int lane = threadIdx.x & 31;
  const int r = rows[eidx];
  const int t = cols[eidx];
  const float w = norm[eidx];
  constexpr int VPT = COUT / 32;  // 4 for 128, 2 for 64
  const float* src = H + (size_t)r * COUT + lane * VPT;
  float* dst = out + (size_t)t * COUT + lane * VPT;
#pragma unroll
  for (int i = 0; i < VPT; ++i) {
    unsafeAtomicAdd(dst + i, src[i] * w);
  }
}

// ---------------- leaky relu (in place) ----------------

__global__ __launch_bounds__(256) void k_leaky(float* __restrict__ x, int n) {
  int i = blockIdx.x * blockDim.x + threadIdx.x;
  if (i < n) { float v = x[i]; x[i] = (v > 0.0f) ? v : NEG_SLOPE * v; }
}

// ---------------------------------------------------------------------------

extern "C" void kernel_launch(void* const* d_in, const int* in_sizes, int n_in,
                              void* d_out, int out_size, void* d_ws, size_t ws_size,
                              hipStream_t stream) {
  const float* x  = (const float*)d_in[0];
  const int*   ei = (const int*)d_in[1];
  const float* ew = (const float*)d_in[2];
  const float* Wm[4] = {(const float*)d_in[3], (const float*)d_in[5],
                        (const float*)d_in[7], (const float*)d_in[9]};
  const float* bm[4] = {(const float*)d_in[4], (const float*)d_in[6],
                        (const float*)d_in[8], (const float*)d_in[10]};

  const int N = in_sizes[0] / 128;
  const int E = in_sizes[2];
  const int* rows = ei;       // edge_index[0] = source
  const int* cols = ei + E;   // edge_index[1] = target

  // workspace carve-out (256B aligned)
  char* p = (char*)d_ws;
  auto carve = [&](size_t bytes) -> float* {
    float* r = (float*)p;
    p += (bytes + 255) & ~(size_t)255;
    return r;
  };
  float* norm = carve((size_t)E * 4);
  float* dinv = carve((size_t)N * 4);              // degree -> dinv in place
  float* hbuf = carve((size_t)N * 128 * 4);
  float* bufA = carve((size_t)N * 128 * 4);
  float* bufB = carve((size_t)N * 128 * 4);
  float* out  = (float*)d_out;

  const int TB = 256;
  // ---- gcn_norm (once, shared by all layers) ----
  k_deg_init<<<(N + TB - 1) / TB, TB, 0, stream>>>(dinv, N);
  k_deg_acc<<<(E + TB - 1) / TB, TB, 0, stream>>>(cols, ew, dinv, E);
  k_dinv<<<(N + TB - 1) / TB, TB, 0, stream>>>(dinv, N);
  k_norm<<<(E + TB - 1) / TB, TB, 0, stream>>>(rows, cols, ew, dinv, norm, E);

  const int tiles128 = (N / 16) * (128 / 16);
  const int tiles64  = (N / 16) * (64 / 16);
  const int scatter_blks = (E + 7) / 8;

  // ---- layer 1: x -> bufA ----
  k_gemm_wmma<128><<<(tiles128 + 3) / 4, 128, 0, stream>>>(x, Wm[0], hbuf, N);
  k_init_out<128><<<(N * 128 + TB - 1) / TB, TB, 0, stream>>>(hbuf, dinv, bm[0], bufA, N);
  k_scatter<128><<<scatter_blks, TB, 0, stream>>>(rows, cols, norm, hbuf, bufA, E);
  k_leaky<<<(N * 128 + TB - 1) / TB, TB, 0, stream>>>(bufA, N * 128);

  // ---- layer 2: bufA -> bufB ----
  k_gemm_wmma<128><<<(tiles128 + 3) / 4, 128, 0, stream>>>(bufA, Wm[1], hbuf, N);
  k_init_out<128><<<(N * 128 + TB - 1) / TB, TB, 0, stream>>>(hbuf, dinv, bm[1], bufB, N);
  k_scatter<128><<<scatter_blks, TB, 0, stream>>>(rows, cols, norm, hbuf, bufB, E);
  k_leaky<<<(N * 128 + TB - 1) / TB, TB, 0, stream>>>(bufB, N * 128);

  // ---- layer 3: bufB -> bufA ----
  k_gemm_wmma<128><<<(tiles128 + 3) / 4, 128, 0, stream>>>(bufB, Wm[2], hbuf, N);
  k_init_out<128><<<(N * 128 + TB - 1) / TB, TB, 0, stream>>>(hbuf, dinv, bm[2], bufA, N);
  k_scatter<128><<<scatter_blks, TB, 0, stream>>>(rows, cols, norm, hbuf, bufA, E);
  k_leaky<<<(N * 128 + TB - 1) / TB, TB, 0, stream>>>(bufA, N * 128);

  // ---- layer 4: bufA -> d_out (COUT = 64, no activation) ----
  k_gemm_wmma<64><<<(tiles64 + 3) / 4, 128, 0, stream>>>(bufA, Wm[3], hbuf, N);
  k_init_out<64><<<(N * 64 + TB - 1) / TB, TB, 0, stream>>>(hbuf, dinv, bm[3], out, N);
  k_scatter<64><<<scatter_blks, TB, 0, stream>>>(rows, cols, norm, hbuf, out, E);
}
